// DecoderCrossAttention_28784870817994
// MI455X (gfx1250) — matrix operand order repaired and checked by reading, not compiled, not run on hardware
//
#include <hip/hip_runtime.h>
#include <math.h>

// ---------------------------------------------------------------------------
// DecoderCrossAttention on MI455X (gfx1250)
//   * all six GEMMs via v_wmma_f32_16x16x32_bf16 (f32 accumulate)
//   * weight strips staged into LDS with GLOBAL_LOAD_ASYNC_TO_LDS_B128
//     (ASYNCcnt / s_wait_asynccnt), double-buffered
// ---------------------------------------------------------------------------

typedef __attribute__((ext_vector_type(16))) __bf16 v16bf;
typedef __attribute__((ext_vector_type(8)))  __bf16 v8bf;
typedef __attribute__((ext_vector_type(8)))  float  v8f;

#define EMBD 1024
#define COND 768
#define NHEAD 16
#define HDIM 64
#define BATCH 8
#define SDIM 2048
#define TDIM 1024

__device__ __forceinline__ v16bf cat16(v8bf lo, v8bf hi) {
  return __builtin_shufflevector(lo, hi, 0, 1, 2, 3, 4, 5, 6, 7, 8, 9, 10, 11, 12, 13, 14, 15);
}

// A-matrix fragment (16x32 bf16, row-major source, stride lda elements).
// Lane l: m = l&15, half = l>>4. elems e -> k = kbase + half*16 + e (contiguous 32B).
__device__ __forceinline__ v16bf load_frag_a(const __bf16* A, int lda, int kbase) {
  int lane = threadIdx.x & 31;
  int m = lane & 15, half = lane >> 4;
  const __bf16* p = A + (size_t)m * lda + kbase + half * 16;
  v8bf a0 = *(const v8bf*)p;
  v8bf a1 = *(const v8bf*)(p + 8);
  return cat16(a0, a1);
}

// B-matrix fragment (32x16 bf16) loaded from Bt (row-major [N][K] = B transposed).
// Lane l: n = l&15, half = l>>4; elems e -> k = kbase + half*16 + e (contiguous 32B).
__device__ __forceinline__ v16bf load_frag_b(const __bf16* Bt, int ldb, int colbase, int kbase) {
  int lane = threadIdx.x & 31;
  int n = lane & 15, half = lane >> 4;
  const __bf16* p = Bt + (size_t)(colbase + n) * ldb + kbase + half * 16;
  v8bf b0 = *(const v8bf*)p;
  v8bf b1 = *(const v8bf*)(p + 8);
  return cat16(b0, b1);
}

__device__ __forceinline__ v8f wmma_bf16(v16bf a, v16bf b, v8f c) {
  return __builtin_amdgcn_wmma_f32_16x16x32_bf16(false, a, false, b, (short)0, c, false, false);
}

// Async global -> LDS copy of 16 bytes (per lane), tracked by ASYNCcnt.
__device__ __forceinline__ void async_copy_b128(const __bf16* gsrc, void* lds_dst) {
  unsigned int loff = (unsigned int)(uintptr_t)lds_dst;     // low 32 bits = LDS byte addr
  unsigned long long gaddr = (unsigned long long)(uintptr_t)gsrc;
  asm volatile("global_load_async_to_lds_b128 %0, %1, off"
               :
               : "v"(loff), "v"(gaddr)
               : "memory");
}

__device__ __forceinline__ void wait_asynccnt0() {
  asm volatile("s_wait_asynccnt 0x0" ::: "memory");
}

// ---------------------------------------------------------------------------
// Utility kernels
// ---------------------------------------------------------------------------
__global__ void cvt_bf16_kernel(const float* __restrict__ in, __bf16* __restrict__ out, int n) {
  int i = blockIdx.x * blockDim.x + threadIdx.x;
  if (i < n) out[i] = (__bf16)in[i];
}

// in [rows][cols] f32  ->  out [cols][rows] bf16
__global__ void transpose_bf16_kernel(const float* __restrict__ in, __bf16* __restrict__ out,
                                      int rows, int cols) {
  int i = blockIdx.x * blockDim.x + threadIdx.x;
  if (i < rows * cols) {
    int r = i / cols, c = i - r * cols;
    out[(size_t)c * rows + r] = (__bf16)in[i];
  }
}

__global__ void zero_f32_kernel(float* __restrict__ p, int n) {
  int i = blockIdx.x * blockDim.x + threadIdx.x;
  if (i < n) p[i] = 0.0f;
}

// ---------------------------------------------------------------------------
// bf16 WMMA GEMM:  C[M,N] = A[M,K] @ Bt[N,K]^T + bias[N]
// Block = 256 threads = 8 waves. Block tile 128x64; wave tile 16x64
// (4 accumulators -> one A fragment feeds 4 WMMAs).
// The 64-wide Bt strip is double-buffered in LDS via async global->LDS loads.
// mode 0: f32 row-major [M,N]         (output projection)
// mode 1: bf16 per-head [B,H,rows,64]   (Q, K)
// mode 2: bf16 per-head [B,H,64,rows]   (V transposed)
// ---------------------------------------------------------------------------
#define BK 32
#define BT_STRIDE 48  // padded LDS row stride (elements): 96B keeps 16B alignment

__global__ void gemm_bf16_kernel(const __bf16* __restrict__ A, const __bf16* __restrict__ Bt,
                                 const float* __restrict__ bias, void* __restrict__ out,
                                 int M, int N, int K, int mode, int rows_per_batch) {
  __shared__ __bf16 bt_lds[2][64][BT_STRIDE];  // 2 x 64 rows x 32 k (padded) = 12 KB

  int wave = threadIdx.x >> 5;
  int lane = threadIdx.x & 31;
  int row_base = blockIdx.x * 128 + wave * 16;
  int col_base = blockIdx.y * 64;

  const __bf16* Arow = A + (size_t)row_base * K;

  // cooperative B-strip loader: thread t moves 16B; n = t/4, 8-element chunk = t%4
  int tn = threadIdx.x >> 2, tc = threadIdx.x & 3;
  const __bf16* bsrc = Bt + (size_t)(col_base + tn) * K + tc * 8;

  v8f acc0 = {}, acc1 = {}, acc2 = {}, acc3 = {};

  async_copy_b128(bsrc, &bt_lds[0][tn][tc * 8]);

  int half = lane >> 4, n = lane & 15;
  int buf = 0;
  for (int kb = 0; kb < K; kb += BK) {
    wait_asynccnt0();
    __syncthreads();
    if (kb + BK < K) async_copy_b128(bsrc + (kb + BK), &bt_lds[buf ^ 1][tn][tc * 8]);

    v16bf af = load_frag_a(Arow, K, kb);
#pragma unroll
    for (int nt = 0; nt < 4; ++nt) {
      const __bf16* p = &bt_lds[buf][nt * 16 + n][half * 16];
      v16bf bf_ = cat16(*(const v8bf*)p, *(const v8bf*)(p + 8));
      if (nt == 0) acc0 = wmma_bf16(af, bf_, acc0);
      else if (nt == 1) acc1 = wmma_bf16(af, bf_, acc1);
      else if (nt == 2) acc2 = wmma_bf16(af, bf_, acc2);
      else acc3 = wmma_bf16(af, bf_, acc3);
    }
    buf ^= 1;
  }

  float accs[4][8];
#pragma unroll
  for (int r = 0; r < 8; ++r) {
    accs[0][r] = acc0[r]; accs[1][r] = acc1[r]; accs[2][r] = acc2[r]; accs[3][r] = acc3[r];
  }

#pragma unroll
  for (int nt = 0; nt < 4; ++nt) {
    int col = col_base + nt * 16 + n;
    float bv = bias[col];
#pragma unroll
    for (int r = 0; r < 8; ++r) {
      int m = row_base + r + half * 8;
      float val = accs[nt][r] + bv;
      if (mode == 0) {
        ((float*)out)[(size_t)m * N + col] = val;
      } else {
        int b = m / rows_per_batch, s = m - b * rows_per_batch;
        int h = col >> 6, d = col & 63;
        if (mode == 1)
          ((__bf16*)out)[(((size_t)b * NHEAD + h) * rows_per_batch + s) * HDIM + d] = (__bf16)val;
        else
          ((__bf16*)out)[(((size_t)b * NHEAD + h) * HDIM + d) * rows_per_batch + s] = (__bf16)val;
      }
    }
  }
}

// ---------------------------------------------------------------------------
// Attention: one wave (32 threads) per block handles (b, h, 16 query rows).
//   scores[16,1024] = Q16x64 @ K^T  (128 WMMAs) -> LDS f32
//   softmax in-wave (2 lanes per row), avg_attn accumulated via atomics
//   out16x64 = attn @ V              (128 WMMAs) -> bf16 attn_out [B,S,E]
// ---------------------------------------------------------------------------
__global__ void attention_kernel(const __bf16* __restrict__ qb, const __bf16* __restrict__ kb,
                                 const __bf16* __restrict__ vtb, __bf16* __restrict__ attn_out,
                                 float* __restrict__ avg_out) {
  __shared__ float smem[16 * TDIM];  // 64 KB of the 320 KB WGP LDS

  int bid = blockIdx.x;
  int s_tile = bid & (SDIM / 16 - 1);  // 0..127
  int h = (bid >> 7) & (NHEAD - 1);
  int b = bid >> 11;

  int lane = threadIdx.x & 31;
  int half = lane >> 4, lm = lane & 15;

  const __bf16* Q  = qb  + (((size_t)b * NHEAD + h) * SDIM + s_tile * 16) * HDIM;
  const __bf16* Kh = kb  + (((size_t)b * NHEAD + h) * TDIM) * HDIM;
  const __bf16* Vt = vtb + (((size_t)b * NHEAD + h) * HDIM) * TDIM;

  // ---- phase 1: scores ----
  v16bf qf0 = load_frag_a(Q, HDIM, 0);
  v16bf qf1 = load_frag_a(Q, HDIM, 32);
  const float scale = 0.125f;  // 1/sqrt(64)
  for (int tt = 0; tt < TDIM / 16; ++tt) {
    v8f acc = {};
    acc = wmma_bf16(qf0, load_frag_b(Kh, HDIM, tt * 16, 0), acc);
    acc = wmma_bf16(qf1, load_frag_b(Kh, HDIM, tt * 16, 32), acc);
#pragma unroll
    for (int r = 0; r < 8; ++r)
      smem[(r + half * 8) * TDIM + tt * 16 + lm] = acc[r] * scale;
  }
  __syncthreads();

  // ---- phase 2: softmax (lane pair (lm, lm+16) shares row lm) ----
  float* row = smem + lm * TDIM + half * (TDIM / 2);
  float mx = -1e30f;
  for (int i = 0; i < TDIM / 2; ++i) mx = fmaxf(mx, row[i]);
  mx = fmaxf(mx, __shfl_xor(mx, 16, 32));
  float sum = 0.0f;
  for (int i = 0; i < TDIM / 2; ++i) {
    float e = __expf(row[i] - mx);
    row[i] = e;
    sum += e;
  }
  sum += __shfl_xor(sum, 16, 32);
  float inv = 1.0f / sum;
  float* avg_base = avg_out + ((size_t)b * SDIM + s_tile * 16 + lm) * TDIM + half * (TDIM / 2);
  for (int i = 0; i < TDIM / 2; ++i) {
    float p = row[i] * inv;
    row[i] = p;
    atomicAdd(avg_base + i, p * (1.0f / NHEAD));  // mean over heads
  }
  __syncthreads();

  // ---- phase 3: out = attn @ V ----
  v8f acc0 = {}, acc1 = {}, acc2 = {}, acc3 = {};
  for (int kk = 0; kk < TDIM; kk += 32) {
    // A fragment from LDS f32 -> bf16
    const float* p = smem + (size_t)lm * TDIM + kk + half * 16;
    v16bf af;
#pragma unroll
    for (int e = 0; e < 8; ++e) {
      af[e] = (__bf16)p[e];
      af[e + 8] = (__bf16)p[e + 8];
    }
    acc0 = wmma_bf16(af, load_frag_b(Vt, TDIM, 0, kk), acc0);
    acc1 = wmma_bf16(af, load_frag_b(Vt, TDIM, 16, kk), acc1);
    acc2 = wmma_bf16(af, load_frag_b(Vt, TDIM, 32, kk), acc2);
    acc3 = wmma_bf16(af, load_frag_b(Vt, TDIM, 48, kk), acc3);
  }

  __bf16* outp = attn_out + ((size_t)b * SDIM + s_tile * 16) * EMBD + h * HDIM;
#pragma unroll
  for (int r = 0; r < 8; ++r) {
    int m = r + half * 8;
    outp[(size_t)m * EMBD +  0 + lm] = (__bf16)acc0[r];
    outp[(size_t)m * EMBD + 16 + lm] = (__bf16)acc1[r];
    outp[(size_t)m * EMBD + 32 + lm] = (__bf16)acc2[r];
    outp[(size_t)m * EMBD + 48 + lm] = (__bf16)acc3[r];
  }
}

// ---------------------------------------------------------------------------
// Host launcher
// ---------------------------------------------------------------------------
extern "C" void kernel_launch(void* const* d_in, const int* in_sizes, int n_in,
                              void* d_out, int out_size, void* d_ws, size_t ws_size,
                              hipStream_t stream) {
  const float* x   = (const float*)d_in[0];
  const float* enc = (const float*)d_in[1];
  const float* Wq  = (const float*)d_in[2];
  const float* bq  = (const float*)d_in[3];
  const float* Wk  = (const float*)d_in[4];
  const float* bk  = (const float*)d_in[5];
  const float* Wv  = (const float*)d_in[6];
  const float* bv  = (const float*)d_in[7];
  const float* Wo  = (const float*)d_in[8];
  const float* bo  = (const float*)d_in[9];

  float* out_proj = (float*)d_out;                                // [B,S,E]
  float* avg_attn = (float*)d_out + (size_t)BATCH * SDIM * EMBD;  // [B,S,T]

  // workspace layout (bytes)
  char* ws = (char*)d_ws;
  size_t off = 0;
  auto alloc = [&](size_t bytes) {
    char* p = ws + off;
    off += (bytes + 255) & ~(size_t)255;
    return p;
  };
  __bf16* xb   = (__bf16*)alloc((size_t)BATCH * SDIM * EMBD * 2);   // 32 MB
  __bf16* encb = (__bf16*)alloc((size_t)BATCH * TDIM * COND * 2);   // 12 MB
  __bf16* WqT  = (__bf16*)alloc((size_t)EMBD * EMBD * 2);
  __bf16* WkT  = (__bf16*)alloc((size_t)EMBD * COND * 2);
  __bf16* WvT  = (__bf16*)alloc((size_t)EMBD * COND * 2);
  __bf16* WoT  = (__bf16*)alloc((size_t)EMBD * EMBD * 2);
  __bf16* qb   = (__bf16*)alloc((size_t)BATCH * NHEAD * SDIM * HDIM * 2);  // 32 MB
  __bf16* kb   = (__bf16*)alloc((size_t)BATCH * NHEAD * TDIM * HDIM * 2);  // 16 MB
  __bf16* vtb  = (__bf16*)alloc((size_t)BATCH * NHEAD * HDIM * TDIM * 2);  // 16 MB
  __bf16* aout = (__bf16*)alloc((size_t)BATCH * SDIM * EMBD * 2);          // 32 MB

  const int nx = BATCH * SDIM * EMBD;       // 16777216
  const int ne = BATCH * TDIM * COND;       // 6291456
  const int navg = BATCH * SDIM * TDIM;     // 16777216

  // 1) convert activations to bf16
  cvt_bf16_kernel<<<(nx + 255) / 256, 256, 0, stream>>>(x, xb, nx);
  cvt_bf16_kernel<<<(ne + 255) / 256, 256, 0, stream>>>(enc, encb, ne);

  // 2) transpose + convert weights (Bt layout = [N][K])
  transpose_bf16_kernel<<<(EMBD * EMBD + 255) / 256, 256, 0, stream>>>(Wq, WqT, EMBD, EMBD);
  transpose_bf16_kernel<<<(COND * EMBD + 255) / 256, 256, 0, stream>>>(Wk, WkT, COND, EMBD);
  transpose_bf16_kernel<<<(COND * EMBD + 255) / 256, 256, 0, stream>>>(Wv, WvT, COND, EMBD);
  transpose_bf16_kernel<<<(EMBD * EMBD + 255) / 256, 256, 0, stream>>>(Wo, WoT, EMBD, EMBD);

  // 3) projections (Q,K per-head; V per-head transposed)
  {
    dim3 g(BATCH * SDIM / 128, EMBD / 64);
    gemm_bf16_kernel<<<g, 256, 0, stream>>>(xb, WqT, bq, qb, BATCH * SDIM, EMBD, EMBD, 1, SDIM);
  }
  {
    dim3 g(BATCH * TDIM / 128, EMBD / 64);
    gemm_bf16_kernel<<<g, 256, 0, stream>>>(encb, WkT, bk, kb, BATCH * TDIM, EMBD, COND, 1, TDIM);
    gemm_bf16_kernel<<<g, 256, 0, stream>>>(encb, WvT, bv, vtb, BATCH * TDIM, EMBD, COND, 2, TDIM);
  }

  // 4) zero avg_attn (accumulated with atomics; d_out is poisoned by harness)
  zero_f32_kernel<<<(navg + 255) / 256, 256, 0, stream>>>(avg_attn, navg);

  // 5) attention
  attention_kernel<<<BATCH * NHEAD * (SDIM / 16), 32, 0, stream>>>(qb, kb, vtb, aout, avg_attn);

  // 6) output projection -> f32 d_out
  {
    dim3 g(BATCH * SDIM / 128, EMBD / 64);
    gemm_bf16_kernel<<<g, 256, 0, stream>>>(aout, WoT, bo, out_proj, BATCH * SDIM, EMBD, EMBD, 0, SDIM);
  }
}
